// edge_function_1_21337397526631
// MI455X (gfx1250) — matrix operand order repaired and checked by reading, not compiled, not run on hardware
//
#include <hip/hip_runtime.h>

// CDNA5 / gfx1250: wave32, WMMA bf16 16x16x32 with f32 accumulate.
// bf16x3 split (hi*hi + hi*lo + lo*hi) recovers ~fp32 accuracy while keeping
// the kernel under the 36us HBM roofline (819MB x-read @ 23.3TB/s).
typedef __attribute__((ext_vector_type(16))) __bf16 v16bf;
typedef __attribute__((ext_vector_type(2)))  __bf16 v2bf;
typedef __attribute__((ext_vector_type(8)))  float  v8f;
typedef __attribute__((ext_vector_type(4)))  float  v4f;
typedef __attribute__((ext_vector_type(2)))  float  v2f;

#define N_NODES 100000
#define K_NEIGH 32
#define N_IN    64
#define N_OUT   64

// ---------------------------------------------------------------------------
// Pre-pass: split W (fp32 [k][n]) into bf16 hi/lo, TRANSPOSED to [n][k] so the
// GEMM kernel's B-fragments (16 consecutive k at fixed n) are contiguous.
// ---------------------------------------------------------------------------
__global__ __launch_bounds__(256) void w_split_kernel(
    const float* __restrict__ W,      // [N_IN, N_OUT]
    __bf16* __restrict__ whiT,        // [N_OUT, N_IN]
    __bf16* __restrict__ wloT)        // [N_OUT, N_IN]
{
    const int idx = blockIdx.x * blockDim.x + threadIdx.x;   // 0..4095
    const int k = idx >> 6;
    const int n = idx & 63;
    const float w = W[idx];                                  // W[k*64 + n]
    const __bf16 h = (__bf16)w;
    whiT[n * N_IN + k] = h;
    wloT[n * N_IN + k] = (__bf16)(w - (float)h);
}

// ---------------------------------------------------------------------------
// Main: one wave per node (grid-stride). 32 edges = 2 M-tiles of 16; N=64 =
// 4 col-tiles; K=64 = 2 chunks of 32 -> 2*4*2*3 = 48 v_wmma per node.
// ---------------------------------------------------------------------------
__global__ __launch_bounds__(256) void gnn_edge_mlp_maxpool(
    const float*  __restrict__ x,     // [N_NODES, K_NEIGH, N_IN]
    const __bf16* __restrict__ whiT,  // [N_OUT, N_IN]
    const __bf16* __restrict__ wloT,  // [N_OUT, N_IN]
    const float*  __restrict__ b,     // [N_OUT]
    float* __restrict__ out)          // [N_NODES, N_OUT]
{
    const int lane    = threadIdx.x & 31;
    const int l15     = lane & 15;
    const int halfsel = lane >> 4;               // 0: lanes 0-15, 1: lanes 16-31

    const int wavesPerBlock = blockDim.x >> 5;
    const int waveId = blockIdx.x * wavesPerBlock + (threadIdx.x >> 5);
    const int nWaves = gridDim.x * wavesPerBlock;

    // ---- B fragments: direct contiguous bf16 loads, no conversion VALU ----
    // B 32x16 bf16, contiguous k order used here (matches A): col n = lane&15; lanes 0-15 hold K=0..15
    // (elem e -> K=e), lanes 16-31 hold K=16..31.
    v16bf Whi[4][2], Wlo[4][2];
    #pragma unroll
    for (int ct = 0; ct < 4; ++ct) {
        const int n = ct * 16 + l15;
        #pragma unroll
        for (int kc = 0; kc < 2; ++kc) {
            const int kb = kc * 32 + halfsel * 16;
            Whi[ct][kc] = *(const v16bf*)(whiT + n * N_IN + kb);  // 32B
            Wlo[ct][kc] = *(const v16bf*)(wloT + n * N_IN + kb);
        }
    }

    float bias[4];
    #pragma unroll
    for (int ct = 0; ct < 4; ++ct) bias[ct] = b[ct * 16 + l15];

    for (int node = waveId; node < N_NODES; node += nWaves) {
        const float* xn = x + (size_t)node * (K_NEIGH * N_IN);

        // Software prefetch of the next node's x block (global_prefetch_b8):
        // 32 lanes x 256B covers the 8KB tile.
        const int nnext = node + nWaves;
        if (nnext < N_NODES) {
            const char* p = (const char*)(x + (size_t)nnext * (K_NEIGH * N_IN));
            __builtin_prefetch(p + lane * 256, 0, 3);
            __builtin_prefetch(p + lane * 256 + 128, 0, 3);
        }

        // ---- Hoist ALL raw f32 loads for this node (16x global_load_b128)
        // so the whole node's traffic is in flight before conversions start.
        // A 16x32 bf16, contiguous k order used here (matches B): row = lane&15;
        //   lane half 0: K = 0..15; half 1: K = 16..31
        float t[2][2][16];
        #pragma unroll
        for (int mh = 0; mh < 2; ++mh) {
            #pragma unroll
            for (int kc = 0; kc < 2; ++kc) {
                const float* xr = xn + (size_t)(mh * 16 + l15) * N_IN
                                  + kc * 32 + halfsel * 16;
                *(v4f*)&t[mh][kc][0]  = *(const v4f*)(xr + 0);   // K+0..3
                *(v4f*)&t[mh][kc][4]  = *(const v4f*)(xr + 4);   // K+4..7
                *(v4f*)&t[mh][kc][8]  = *(const v4f*)(xr + 8);   // K+8..11
                *(v4f*)&t[mh][kc][12] = *(const v4f*)(xr + 12);  // K+12..15
            }
        }

        // ---- Pairwise bf16 hi/lo split: per element pair this is
        // 1x v_cvt_pk_bf16_f32 (hi) + 2x v_fma_mix (residual) + 1x v_cvt_pk
        // (lo) = 2 VALU/element, co-executing with the WMMA chain.
        v16bf Ahi[2][2], Alo[2][2];
        #pragma unroll
        for (int mh = 0; mh < 2; ++mh) {
            #pragma unroll
            for (int kc = 0; kc < 2; ++kc) {
                #pragma unroll
                for (int e = 0; e < 16; e += 2) {
                    const float x0 = t[mh][kc][e], x1 = t[mh][kc][e + 1];
                    v2f  xp = { x0, x1 };
                    v2bf hp = __builtin_convertvector(xp, v2bf);
                    v2f  rp = { x0 - (float)hp[0], x1 - (float)hp[1] };
                    v2bf lp = __builtin_convertvector(rp, v2bf);
                    Ahi[mh][kc][e]     = hp[0];
                    Ahi[mh][kc][e + 1] = hp[1];
                    Alo[mh][kc][e]     = lp[0];
                    Alo[mh][kc][e + 1] = lp[1];
                }
            }
        }

        // ---- 6 WMMAs per (M-tile, col-tile), starting from inline-0 C ----
        v8f rmax[4];
        #pragma unroll
        for (int mh = 0; mh < 2; ++mh) {
            #pragma unroll
            for (int ct = 0; ct < 4; ++ct) {
                v8f c = {};   // folds to inline 0 C operand
                c = __builtin_amdgcn_wmma_f32_16x16x32_bf16(
                    false, Ahi[mh][0], false, Whi[ct][0], (short)0, c, false, false);
                c = __builtin_amdgcn_wmma_f32_16x16x32_bf16(
                    false, Ahi[mh][0], false, Wlo[ct][0], (short)0, c, false, false);
                c = __builtin_amdgcn_wmma_f32_16x16x32_bf16(
                    false, Alo[mh][0], false, Whi[ct][0], (short)0, c, false, false);
                c = __builtin_amdgcn_wmma_f32_16x16x32_bf16(
                    false, Ahi[mh][1], false, Whi[ct][1], (short)0, c, false, false);
                c = __builtin_amdgcn_wmma_f32_16x16x32_bf16(
                    false, Ahi[mh][1], false, Wlo[ct][1], (short)0, c, false, false);
                c = __builtin_amdgcn_wmma_f32_16x16x32_bf16(
                    false, Alo[mh][1], false, Whi[ct][1], (short)0, c, false, false);
                if (mh == 0) {
                    rmax[ct] = c;                  // first tile: no init needed
                } else {
                    #pragma unroll
                    for (int i = 0; i < 8; ++i)
                        rmax[ct][i] = fmaxf(rmax[ct][i], c[i]);
                }
            }
        }

        // Reduce over M: reg i holds rows i (lanes 0-15) / i+8 (lanes 16-31);
        // pairwise-tree fold, swap lane halves (xor 16), bias + ReLU, store.
        #pragma unroll
        for (int ct = 0; ct < 4; ++ct) {
            float m01 = fmaxf(rmax[ct][0], rmax[ct][1]);
            float m23 = fmaxf(rmax[ct][2], rmax[ct][3]);
            float m45 = fmaxf(rmax[ct][4], rmax[ct][5]);
            float m67 = fmaxf(rmax[ct][6], rmax[ct][7]);
            float m = fmaxf(fmaxf(m01, m23), fmaxf(m45, m67));
            m = fmaxf(m, __shfl_xor(m, 16, 32));
            m = fmaxf(m + bias[ct], 0.0f);        // hoisted bias + ReLU
            if (lane < 16)
                out[(size_t)node * N_OUT + ct * 16 + l15] = m;
        }
    }
}

extern "C" void kernel_launch(void* const* d_in, const int* in_sizes, int n_in,
                              void* d_out, int out_size, void* d_ws, size_t ws_size,
                              hipStream_t stream) {
    (void)in_sizes; (void)n_in; (void)out_size; (void)ws_size;
    const float* x = (const float*)d_in[0];
    const float* W = (const float*)d_in[1];
    const float* b = (const float*)d_in[2];
    float* out = (float*)d_out;

    // Workspace: bf16 hi/lo transposed copies of W (2 * 4096 * 2B = 16KB).
    __bf16* whiT = (__bf16*)d_ws;
    __bf16* wloT = whiT + (N_IN * N_OUT);

    w_split_kernel<<<dim3(16), dim3(256), 0, stream>>>(W, whiT, wloT);

    // 1280 blocks x 8 waves = 10240 waves; ~10 nodes/wave.
    gnn_edge_mlp_maxpool<<<dim3(1280), dim3(256), 0, stream>>>(
        x, whiT, wloT, b, out);
}
